// DecipherModel_17222818857432
// MI455X (gfx1250) — matrix operand order repaired and checked by reading, not compiled, not run on hardware
//
#include <hip/hip_runtime.h>

// ---------------------------------------------------------------------------
// DecipherModel on MI455X (gfx1250): bf16 WMMA GEMM cores + VALU fused epilogues
// Dims (compile-time): B=2048 S=30 T=30 EMB=H=512 4H=2048 L=2 V=64 CHARSET=128
// GEMM core: wave32, V_WMMA_F32_16X16X32_BF16, 32x64 tile per wave
// ---------------------------------------------------------------------------

typedef __attribute__((ext_vector_type(16))) __bf16 v16bf;
typedef __attribute__((ext_vector_type(8)))  float  v8f;
typedef unsigned short u16;

#define CB 2048   // batch
#define CS 30     // src len
#define CT 30     // tgt len
#define CE 512    // EMB (= H)
#define CH 512    // H
#define CH4 2048  // 4H
#define CV 64     // V_known
#define CCH 128   // charset

__device__ __forceinline__ u16 f2bf(float f) {
  unsigned int x = __float_as_uint(f);
  unsigned int r = (x + 0x7FFFu + ((x >> 16) & 1u)) >> 16;  // round-nearest-even
  return (u16)r;
}
__device__ __forceinline__ float bf2f(u16 h) {
  return __uint_as_float(((unsigned int)h) << 16);
}
__device__ __forceinline__ float sigm(float x) { return 1.f / (1.f + __expf(-x)); }

struct alignas(16) Q16 { unsigned int a, b, c, d; };
union FragBF { v16bf v; Q16 q[2]; };

// ---------------------------------------------------------------------------
// bf16 WMMA GEMM: C[M,N] (f32) = A[M,K](bf16,row-major) @ W[N,K](bf16,row-major)^T
// Block = 128 threads = 4 waves; each wave computes a 32(M) x 64(N) tile
// (2 M sub-tiles x 4 N sub-tiles = 8 WMMAs per K-step, B fragment reused x2).
// Fragment layouts used by this code (wave32, contiguous k, not the ISA table order):
//   A 16x32: lane m(<16) holds row m, K {k0..k0+15}; lane m+16 row m, K {k0+16..k0+31}
//   B 32x16: lane n(<16) holds col n, K {k0..k0+15}; lane n+16 col n, K {k0+16..k0+31}
//   C 16x16: col = lane%16, row = vgpr + 8*(lane>=16)
// ---------------------------------------------------------------------------
__global__ __launch_bounds__(128) void gemm_bf16_wmma(
    const u16* __restrict__ A, int lda, const u16* __restrict__ W, int ldw,
    float* __restrict__ C, int ldc, const float* __restrict__ bias,
    int M, int N, int K, int accum) {
  const int wave = threadIdx.x >> 5;
  const int lane = threadIdx.x & 31;
  const int hsel = lane >> 4;
  const int l16  = lane & 15;
  const int m0 = (blockIdx.x * 4 + wave) * 32;
  const int n0 = blockIdx.y * 64;
  if (m0 >= M) return;  // wave-uniform; EXEC stays all-ones inside WMMA region

  const u16* Arow0 = A + (size_t)(m0 + l16) * lda;
  const u16* Arow1 = A + (size_t)(m0 + 16 + l16) * lda;
  const u16* Wrow0 = W + (size_t)(n0 +  0 + l16) * ldw;
  const u16* Wrow1 = W + (size_t)(n0 + 16 + l16) * ldw;
  const u16* Wrow2 = W + (size_t)(n0 + 32 + l16) * ldw;
  const u16* Wrow3 = W + (size_t)(n0 + 48 + l16) * ldw;

  v8f acc[8];
#pragma unroll
  for (int j = 0; j < 8; ++j) acc[j] = (v8f){0.f, 0.f, 0.f, 0.f, 0.f, 0.f, 0.f, 0.f};

  for (int k0 = 0; k0 < K; k0 += 32) {
    FragBF fa0, fa1;
    fa0.q[0] = *(const Q16*)(Arow0 + k0 + hsel * 16);
    fa0.q[1] = *(const Q16*)(Arow0 + k0 + hsel * 16 + 8);
    fa1.q[0] = *(const Q16*)(Arow1 + k0 + hsel * 16);
    fa1.q[1] = *(const Q16*)(Arow1 + k0 + hsel * 16 + 8);
    __builtin_prefetch(Arow0 + k0 + 64, 0, 1);
    __builtin_prefetch(Arow1 + k0 + 64, 0, 1);

    FragBF fb;
    fb.q[0] = *(const Q16*)(Wrow0 + k0 + hsel * 16);
    fb.q[1] = *(const Q16*)(Wrow0 + k0 + hsel * 16 + 8);
    acc[0] = __builtin_amdgcn_wmma_f32_16x16x32_bf16(false, fa0.v, false, fb.v,
                                                     (short)0, acc[0], false, false);
    acc[4] = __builtin_amdgcn_wmma_f32_16x16x32_bf16(false, fa1.v, false, fb.v,
                                                     (short)0, acc[4], false, false);
    fb.q[0] = *(const Q16*)(Wrow1 + k0 + hsel * 16);
    fb.q[1] = *(const Q16*)(Wrow1 + k0 + hsel * 16 + 8);
    acc[1] = __builtin_amdgcn_wmma_f32_16x16x32_bf16(false, fa0.v, false, fb.v,
                                                     (short)0, acc[1], false, false);
    acc[5] = __builtin_amdgcn_wmma_f32_16x16x32_bf16(false, fa1.v, false, fb.v,
                                                     (short)0, acc[5], false, false);
    fb.q[0] = *(const Q16*)(Wrow2 + k0 + hsel * 16);
    fb.q[1] = *(const Q16*)(Wrow2 + k0 + hsel * 16 + 8);
    acc[2] = __builtin_amdgcn_wmma_f32_16x16x32_bf16(false, fa0.v, false, fb.v,
                                                     (short)0, acc[2], false, false);
    acc[6] = __builtin_amdgcn_wmma_f32_16x16x32_bf16(false, fa1.v, false, fb.v,
                                                     (short)0, acc[6], false, false);
    fb.q[0] = *(const Q16*)(Wrow3 + k0 + hsel * 16);
    fb.q[1] = *(const Q16*)(Wrow3 + k0 + hsel * 16 + 8);
    acc[3] = __builtin_amdgcn_wmma_f32_16x16x32_bf16(false, fa0.v, false, fb.v,
                                                     (short)0, acc[3], false, false);
    acc[7] = __builtin_amdgcn_wmma_f32_16x16x32_bf16(false, fa1.v, false, fb.v,
                                                     (short)0, acc[7], false, false);
  }

#pragma unroll
  for (int half = 0; half < 2; ++half) {
#pragma unroll
    for (int j = 0; j < 4; ++j) {
      const int col = n0 + j * 16 + l16;
      const float bv = bias ? bias[col] : 0.f;
      const v8f a = acc[half * 4 + j];
#pragma unroll
      for (int v = 0; v < 8; ++v) {
        const int row = m0 + half * 16 + v + hsel * 8;
        const size_t idx = (size_t)row * ldc + col;
        float out = a[v] + bv;
        if (accum) out += C[idx];
        C[idx] = out;
      }
    }
  }
}

// ---------------------------------------------------------------------------
// Elementwise / small kernels
// ---------------------------------------------------------------------------
__global__ void fill_f32_kernel(float* p, float v, long long n) {
  long long i = (long long)blockIdx.x * blockDim.x + threadIdx.x;
  long long st = (long long)gridDim.x * blockDim.x;
  for (; i < n; i += st) p[i] = v;
}
__global__ void fill_u16_kernel(u16* p, u16 v, long long n) {
  long long i = (long long)blockIdx.x * blockDim.x + threadIdx.x;
  long long st = (long long)gridDim.x * blockDim.x;
  for (; i < n; i += st) p[i] = v;
}
__global__ void conv_bf16_kernel(const float* __restrict__ s, u16* __restrict__ d, long long n) {
  long long i = (long long)blockIdx.x * blockDim.x + threadIdx.x;
  long long st = (long long)gridDim.x * blockDim.x;
  for (; i < n; i += st) d[i] = f2bf(s[i]);
}
// src (R,C) f32 -> dst (C,R) bf16
__global__ void transpose_conv_kernel(const float* __restrict__ s, u16* __restrict__ d,
                                      int R, int C) {
  int i = blockIdx.x * blockDim.x + threadIdx.x;
  if (i >= R * C) return;
  int r = i / C, c = i % C;
  d[(size_t)c * R + r] = f2bf(s[i]);
}
// E = P @ U  (P: (Vn,CCH), U: (CCH,E))
__global__ void proj_embed_kernel(const float* __restrict__ P, const float* __restrict__ U,
                                  float* __restrict__ Ef, u16* __restrict__ Eb, int Vn, int E) {
  int i = blockIdx.x * blockDim.x + threadIdx.x;
  if (i >= Vn * E) return;
  int v = i / E, e = i % E;
  float s = 0.f;
  for (int c = 0; c < CCH; ++c) s += P[v * CCH + c] * U[(size_t)c * E + e];
  Ef[i] = s;
  if (Eb) Eb[i] = f2bf(s);
}
// emb_b[s][b][e] = bf16(E_lost[id[b][s]][e])
__global__ void gather_emb_kernel(const int* __restrict__ ids, const float* __restrict__ El,
                                  u16* __restrict__ emb) {
  long long i = (long long)blockIdx.x * blockDim.x + threadIdx.x;
  if (i >= (long long)CS * CB * CE) return;
  int e = (int)(i % CE);
  long long sb = i / CE;
  int b = (int)(sb % CB);
  int s = (int)(sb / CB);
  int id = ids[(size_t)b * CS + s];
  emb[i] = f2bf(El[(size_t)id * CE + e]);
}
// LSTM cell epilogue: gates (B,4H) f32 -> h,c; also write bf16 h and optional
// bf16 copy into the (S,B,2H) layer-output slot and optional hT/cT capture.
__global__ void lstm_cell_kernel(const float* __restrict__ gates, float* __restrict__ h,
                                 float* __restrict__ c, u16* __restrict__ h_bf,
                                 u16* __restrict__ xout, int xstride,
                                 float* __restrict__ hT, float* __restrict__ cT,
                                 int Bn, int Hn) {
  int i = blockIdx.x * blockDim.x + threadIdx.x;
  if (i >= Bn * Hn) return;
  int b = i / Hn, j = i % Hn;
  const float* g = gates + (size_t)b * 4 * Hn;
  float gi = g[j], gf = g[Hn + j], gg = g[2 * Hn + j], go = g[3 * Hn + j];
  float cn = sigm(gf) * c[i] + sigm(gi) * tanhf(gg);
  float hn = sigm(go) * tanhf(cn);
  c[i] = cn; h[i] = hn; h_bf[i] = f2bf(hn);
  if (xout) xout[(size_t)b * xstride + j] = f2bf(hn);
  if (hT) { hT[i] = hn; cT[i] = cn; }
}
// decoder init: h_tilde = 0, inp_emb = bf16(E_known[SOW_ID])
__global__ void dec_init_kernel(const float* __restrict__ Ek, float* __restrict__ ht,
                                u16* __restrict__ ie) {
  int i = blockIdx.x * blockDim.x + threadIdx.x;
  if (i >= CB * CE) return;
  int e = i % CE;
  ht[i] = 0.f;
  ie[i] = f2bf(Ek[(size_t)1 * CE + e]);  // SOW_ID = 1
}
// inp (B,2E) bf16 = [h_tilde f32 | inp_emb bf16]
__global__ void build_inp_kernel(const float* __restrict__ ht, const u16* __restrict__ ie,
                                 u16* __restrict__ inp) {
  int i = blockIdx.x * blockDim.x + threadIdx.x;
  if (i >= CB * 2 * CE) return;
  int b = i / (2 * CE), j = i % (2 * CE);
  inp[i] = (j < CE) ? f2bf(ht[(size_t)b * CE + j]) : ie[(size_t)b * CE + (j - CE)];
}
// attention: one block (256 thr) per batch row. scores -> softmax -> almt out,
// ctx_s (into cat), ctx_t (into cat), ctx_s_emb.
__global__ __launch_bounds__(256) void attn_kernel(
    const float* __restrict__ hsWa,   // (S,B,H)
    const float* __restrict__ ctx_t,  // (B,H)
    const u16* __restrict__ hs_bf,    // (S,B,2H)
    const u16* __restrict__ emb_bf,   // (S,B,E)
    float* __restrict__ almt_out,     // base of almt output, (B,T,S)
    u16* __restrict__ cat_bf,         // (B,3H)
    float* __restrict__ cse,          // (B,E)
    int t) {
  const int b = blockIdx.x;
  const int tid = threadIdx.x, w = tid >> 5, ln = tid & 31;
  __shared__ float sc[32];
  const float* crow = ctx_t + (size_t)b * CH;
  for (int s = w; s < CS; s += 8) {
    const float* row = hsWa + ((size_t)s * CB + b) * CH;
    float acc = 0.f;
    for (int k = ln; k < CH; k += 32) acc += row[k] * crow[k];
    for (int o = 16; o; o >>= 1) acc += __shfl_xor(acc, o);
    if (ln == 0) sc[s] = acc;
  }
  __syncthreads();
  if (tid < 32) {
    float v = (ln < CS) ? sc[ln] : -3.0e38f;
    float m = v;
    for (int o = 16; o; o >>= 1) m = fmaxf(m, __shfl_xor(m, o));
    float e = (ln < CS) ? __expf(v - m) : 0.f;
    float sum = e;
    for (int o = 16; o; o >>= 1) sum += __shfl_xor(sum, o);
    float a = e / sum;
    if (ln < CS) {
      sc[ln] = a;
      almt_out[(size_t)b * CT * CS + (size_t)t * CS + ln] = a;
    }
  }
  __syncthreads();
  const int H2 = 2 * CH;
  for (int d = tid; d < H2; d += blockDim.x) {
    float acc = 0.f;
#pragma unroll 5
    for (int s = 0; s < CS; ++s) acc += sc[s] * bf2f(hs_bf[((size_t)s * CB + b) * H2 + d]);
    cat_bf[(size_t)b * 3 * CH + d] = f2bf(acc);
  }
  for (int d = tid; d < CH; d += blockDim.x)
    cat_bf[(size_t)b * 3 * CH + H2 + d] = f2bf(crow[d]);
  for (int e = tid; e < CE; e += blockDim.x) {
    float acc = 0.f;
#pragma unroll 5
    for (int s = 0; s < CS; ++s) acc += sc[s] * bf2f(emb_bf[((size_t)s * CB + b) * CE + e]);
    cse[(size_t)b * CE + e] = acc;
  }
}
// leaky_relu + norm-controlled residual: block (128 thr) per batch row
__global__ __launch_bounds__(128) void finalize_kernel(
    float* __restrict__ htr, const float* __restrict__ cse,
    float* __restrict__ ht, u16* __restrict__ ht_bf) {
  const int b = blockIdx.x, tid = threadIdx.x;
  float ss1 = 0.f, ss2 = 0.f;
  for (int e = tid; e < CE; e += blockDim.x) {
    float x = htr[(size_t)b * CE + e];
    x = x > 0.f ? x : 0.01f * x;  // leaky_relu
    htr[(size_t)b * CE + e] = x;
    ss2 += x * x;
    float y = cse[(size_t)b * CE + e];
    ss1 += y * y;
  }
  for (int o = 16; o; o >>= 1) { ss1 += __shfl_xor(ss1, o); ss2 += __shfl_xor(ss2, o); }
  __shared__ float s1[4], s2[4];
  int w = tid >> 5, ln = tid & 31;
  if (ln == 0) { s1[w] = ss1; s2[w] = ss2; }
  __syncthreads();
  float n1 = sqrtf(s1[0] + s1[1] + s1[2] + s1[3]);
  float n2 = sqrtf(s2[0] + s2[1] + s2[2] + s2[3]);
  float k1 = 1.0f * n1 / (n1 + 1e-8f);  // RATIOS[0] = 1.0 for ctx_s_emb
  float k2 = 0.2f * n2 / (n2 + 1e-8f);  // RATIOS[1] = 0.2 for h_tilde_rnn
  for (int e = tid; e < CE; e += blockDim.x) {
    float v = cse[(size_t)b * CE + e] * k1 + htr[(size_t)b * CE + e] * k2;
    ht[(size_t)b * CE + e] = v;
    ht_bf[(size_t)b * CE + e] = f2bf(v);
  }
}
// log_softmax over V=64 + soft embedding; block (128 thr) per batch row
__global__ __launch_bounds__(128) void out_softmax_kernel(
    const float* __restrict__ logits, const float* __restrict__ Ek,
    float* __restrict__ out_lp /* + t*V*B */, u16* __restrict__ inp_emb) {
  const int b = blockIdx.x, tid = threadIdx.x;
  __shared__ float lg[CV], pp[CV], lse;
  if (tid < CV) lg[tid] = logits[(size_t)b * CV + tid];
  __syncthreads();
  if (tid < 32) {
    float a = lg[tid], c = lg[tid + 32];
    float m = fmaxf(a, c);
    for (int o = 16; o; o >>= 1) m = fmaxf(m, __shfl_xor(m, o));
    float s = __expf(a - m) + __expf(c - m);
    for (int o = 16; o; o >>= 1) s += __shfl_xor(s, o);
    if (tid == 0) lse = m + __logf(s);
  }
  __syncthreads();
  if (tid < CV) {
    float lp = lg[tid] - lse;
    out_lp[(size_t)tid * CB + b] = lp;  // out[t][v][b]
    pp[tid] = __expf(lp);
  }
  __syncthreads();
  for (int e = tid; e < CE; e += blockDim.x) {
    float acc = 0.f;
#pragma unroll 8
    for (int v = 0; v < CV; ++v) acc += pp[v] * Ek[(size_t)v * CE + e];
    inp_emb[(size_t)b * CE + e] = f2bf(acc);
  }
}

// ---------------------------------------------------------------------------
// Host orchestration (all launches on `stream`; graph-capture safe)
// Input order assumes recursive insertion-order flattening of setup_inputs().
// ---------------------------------------------------------------------------
extern "C" void kernel_launch(void* const* d_in, const int* in_sizes, int n_in,
                              void* d_out, int out_size, void* d_ws, size_t ws_size,
                              hipStream_t stream) {
  (void)in_sizes; (void)n_in; (void)out_size; (void)ws_size;
  int i = 0;
  const int* id_seqs = (const int*)d_in[i++];   // (B,S)
  i++;                                          // max_len scalar (== CT)
  const float* U       = (const float*)d_in[i++];
  const float* P_lost  = (const float*)d_in[i++];
  const float* P_known = (const float*)d_in[i++];
  const float *eWih[2][2], *eWhh[2][2], *eB[2][2];
  for (int l = 0; l < 2; ++l)
    for (int d = 0; d < 2; ++d) {
      eWih[l][d] = (const float*)d_in[i++];
      eWhh[l][d] = (const float*)d_in[i++];
      eB[l][d]   = (const float*)d_in[i++];
    }
  const float *dWih[2], *dWhh[2], *dB[2];
  for (int l = 0; l < 2; ++l) {
    dWih[l] = (const float*)d_in[i++];
    dWhh[l] = (const float*)d_in[i++];
    dB[l]   = (const float*)d_in[i++];
  }
  const float* Wa = (const float*)d_in[i++];
  const float* Wh = (const float*)d_in[i++];
  const float* bh = (const float*)d_in[i++];

  float* out_lp   = (float*)d_out;
  float* out_almt = out_lp + (size_t)CT * CV * CB;

  // ---- workspace bump allocator ----
  char* ws = (char*)d_ws;
  size_t off = 0;
  auto alloc = [&](size_t bytes) -> void* {
    off = (off + 255) & ~(size_t)255;
    void* p = ws + off;
    off += bytes;
    return p;
  };
  float* E_lost_f  = (float*)alloc(sizeof(float) * CV * CE);  // V_LOST == 64 too
  float* E_known_f = (float*)alloc(sizeof(float) * CV * CE);
  u16*   E_known_b = (u16*)alloc(sizeof(u16) * CV * CE);
  u16*   emb_b = (u16*)alloc(sizeof(u16) * (size_t)CS * CB * CE);
  u16*   xA    = (u16*)alloc(sizeof(u16) * (size_t)CS * CB * 2 * CH);
  u16*   xB    = (u16*)alloc(sizeof(u16) * (size_t)CS * CB * 2 * CH);
  float* hsWa  = (float*)alloc(sizeof(float) * (size_t)CS * CB * CH);
  float* gates = (float*)alloc(sizeof(float) * (size_t)CB * CH4);
  float* h_f   = (float*)alloc(sizeof(float) * (size_t)CB * CH);
  float* c_f   = (float*)alloc(sizeof(float) * (size_t)CB * CH);
  u16*   h_b   = (u16*)alloc(sizeof(u16) * (size_t)CB * CH);
  float* h_dec = (float*)alloc(sizeof(float) * 2 * (size_t)CB * CH);
  float* c_dec = (float*)alloc(sizeof(float) * 2 * (size_t)CB * CH);
  u16*   h_dec_b = (u16*)alloc(sizeof(u16) * 2 * (size_t)CB * CH);
  u16* wEih[2][2], *wEhh[2][2];
  for (int l = 0; l < 2; ++l)
    for (int d = 0; d < 2; ++d) {
      int din = l ? 2 * CH : CE;
      wEih[l][d] = (u16*)alloc(sizeof(u16) * (size_t)CH4 * din);
      wEhh[l][d] = (u16*)alloc(sizeof(u16) * (size_t)CH4 * CH);
    }
  u16* wDih[2], *wDhh[2];
  for (int l = 0; l < 2; ++l) {
    int din = l ? CH : (CE + CH);
    wDih[l] = (u16*)alloc(sizeof(u16) * (size_t)CH4 * din);
    wDhh[l] = (u16*)alloc(sizeof(u16) * (size_t)CH4 * CH);
  }
  u16* WaT_b = (u16*)alloc(sizeof(u16) * (size_t)CH * 2 * CH);   // (H,2H) = Wa^T
  u16* Wh_b  = (u16*)alloc(sizeof(u16) * (size_t)CE * 3 * CH);
  u16* inp_b = (u16*)alloc(sizeof(u16) * (size_t)CB * (CE + CH));
  u16* cat_b = (u16*)alloc(sizeof(u16) * (size_t)CB * 3 * CH);
  float* cse = (float*)alloc(sizeof(float) * (size_t)CB * CE);
  float* htr = (float*)alloc(sizeof(float) * (size_t)CB * CE);
  float* ht  = (float*)alloc(sizeof(float) * (size_t)CB * CE);
  u16*   ht_b = (u16*)alloc(sizeof(u16) * (size_t)CB * CE);
  u16*   ie_b = (u16*)alloc(sizeof(u16) * (size_t)CB * CE);
  float* logits = (float*)alloc(sizeof(float) * (size_t)CB * CV);

  auto conv = [&](const float* s, u16* d, long long n) {
    conv_bf16_kernel<<<dim3((unsigned)((n + 255) / 256)), 256, 0, stream>>>(s, d, n);
  };
  auto gemm = [&](const u16* A, int lda, const u16* W, int ldw, float* C, int ldc,
                  const float* bias, int M, int N, int K, int accum) {
    dim3 g(M / 128, N / 64);  // 4 waves * 32 rows per block; M % 128 == 0 for all calls
    gemm_bf16_wmma<<<g, 128, 0, stream>>>(A, lda, W, ldw, C, ldc, bias, M, N, K, accum);
  };

  // ---- prep: embeddings + weight conversions ----
  proj_embed_kernel<<<(CV * CE + 255) / 256, 256, 0, stream>>>(P_lost, U, E_lost_f, nullptr, CV, CE);
  proj_embed_kernel<<<(CV * CE + 255) / 256, 256, 0, stream>>>(P_known, U, E_known_f, E_known_b, CV, CE);
  gather_emb_kernel<<<(unsigned)(((long long)CS * CB * CE + 255) / 256), 256, 0, stream>>>(
      id_seqs, E_lost_f, emb_b);
  for (int l = 0; l < 2; ++l)
    for (int d = 0; d < 2; ++d) {
      int din = l ? 2 * CH : CE;
      conv(eWih[l][d], wEih[l][d], (long long)CH4 * din);
      conv(eWhh[l][d], wEhh[l][d], (long long)CH4 * CH);
    }
  for (int l = 0; l < 2; ++l) {
    int din = l ? CH : (CE + CH);
    conv(dWih[l], wDih[l], (long long)CH4 * din);
    conv(dWhh[l], wDhh[l], (long long)CH4 * CH);
  }
  transpose_conv_kernel<<<(2 * CH * CH + 255) / 256, 256, 0, stream>>>(Wa, WaT_b, 2 * CH, CH);
  conv(Wh, Wh_b, (long long)CE * 3 * CH);

  // ---- encoder: 2 layers x {fwd,bwd} x S timesteps ----
  for (int l = 0; l < 2; ++l) {
    const u16* lin = (l == 0) ? emb_b : xA;
    int din = (l == 0) ? CE : 2 * CH;
    u16* lout = (l == 0) ? xA : xB;
    for (int d = 0; d < 2; ++d) {
      fill_f32_kernel<<<256, 256, 0, stream>>>(c_f, 0.f, (long long)CB * CH);
      fill_u16_kernel<<<256, 256, 0, stream>>>(h_b, 0, (long long)CB * CH);
      for (int t = 0; t < CS; ++t) {
        int pos = d ? (CS - 1 - t) : t;
        gemm(lin + (size_t)pos * CB * din, din, wEih[l][d], din, gates, CH4,
             eB[l][d], CB, CH4, din, 0);
        gemm(h_b, CH, wEhh[l][d], CH, gates, CH4, nullptr, CB, CH4, CH, 1);
        bool lastf = (d == 0 && t == CS - 1);
        lstm_cell_kernel<<<(CB * CH + 255) / 256, 256, 0, stream>>>(
            gates, h_f, c_f, h_b, lout + (size_t)pos * CB * 2 * CH + (size_t)d * CH,
            2 * CH, lastf ? h_dec + (size_t)l * CB * CH : nullptr,
            lastf ? c_dec + (size_t)l * CB * CH : nullptr, CB, CH);
      }
    }
  }

  // ---- attention precompute: hsWa = h_s @ Wa  (M = S*B = 61440) ----
  gemm(xB, 2 * CH, WaT_b, 2 * CH, hsWa, CH, nullptr, CS * CB, CH, 2 * CH, 0);

  // ---- decoder init ----
  conv(h_dec, h_dec_b, 2LL * CB * CH);
  dec_init_kernel<<<(CB * CE + 255) / 256, 256, 0, stream>>>(E_known_f, ht, ie_b);

  // ---- autoregressive decoder ----
  for (int t = 0; t < CT; ++t) {
    build_inp_kernel<<<(CB * 2 * CE + 255) / 256, 256, 0, stream>>>(ht, ie_b, inp_b);
    // layer 0: inp (B,1024)
    gemm(inp_b, CE + CH, wDih[0], CE + CH, gates, CH4, dB[0], CB, CH4, CE + CH, 0);
    gemm(h_dec_b, CH, wDhh[0], CH, gates, CH4, nullptr, CB, CH4, CH, 1);
    lstm_cell_kernel<<<(CB * CH + 255) / 256, 256, 0, stream>>>(
        gates, h_dec, c_dec, h_dec_b, nullptr, 0, nullptr, nullptr, CB, CH);
    // layer 1: input = layer-0 h
    gemm(h_dec_b, CH, wDih[1], CH, gates, CH4, dB[1], CB, CH4, CH, 0);
    gemm(h_dec_b + (size_t)CB * CH, CH, wDhh[1], CH, gates, CH4, nullptr, CB, CH4, CH, 1);
    lstm_cell_kernel<<<(CB * CH + 255) / 256, 256, 0, stream>>>(
        gates, h_dec + (size_t)CB * CH, c_dec + (size_t)CB * CH,
        h_dec_b + (size_t)CB * CH, nullptr, 0, nullptr, nullptr, CB, CH);
    // attention + context + cat build
    attn_kernel<<<CB, 256, 0, stream>>>(hsWa, h_dec + (size_t)CB * CH, xB, emb_b,
                                        out_almt, cat_b, cse, t);
    // h_tilde_rnn = cat @ Wh^T + bh
    gemm(cat_b, 3 * CH, Wh_b, 3 * CH, htr, CE, bh, CB, CE, 3 * CH, 0);
    finalize_kernel<<<CB, 128, 0, stream>>>(htr, cse, ht, ht_b);
    // logits = h_tilde @ E_known^T
    gemm(ht_b, CE, E_known_b, CE, logits, CV, nullptr, CB, CV, CE, 0);
    out_softmax_kernel<<<CB, 128, 0, stream>>>(logits, E_known_f,
                                               out_lp + (size_t)t * CV * CB, ie_b);
  }
}
